// NODEDecoder_71021579207112
// MI455X (gfx1250) — matrix operand order repaired and checked by reading, not compiled, not run on hardware
//
#include <hip/hip_runtime.h>
#include <cstdint>

#define LATENT 128
#define HIDDEN 256
#define OUTD   8
#define TSTEPS 32
#define BATCH  16384
#define LN_EPS 1e-5f

typedef __attribute__((ext_vector_type(16))) __bf16 v16bf;
typedef __attribute__((ext_vector_type(8)))  float  v8f;

union V16U { unsigned u[8]; v16bf v; };

__device__ __forceinline__ unsigned pk_bf16(float a, float b) {
  union { unsigned u; __bf16 h[2]; } x;
  x.h[0] = (__bf16)a; x.h[1] = (__bf16)b;
  return x.u;
}

__device__ __forceinline__ float leaky(float x) { return fmaxf(x, 0.1f * x); }

__device__ __forceinline__ v8f wmma_bf16(v16bf a, v16bf b, v8f c) {
  // D = A(16x32 bf16) x B(32x16 bf16) + C(f32)
  return __builtin_amdgcn_wmma_f32_16x16x32_bf16(false, a, false, b, (short)0, c,
                                                 false, false);
}

// ---------------------------------------------------------------------------
// Per-wave LDS staging buffers (one 16-row batch tile per wave)
// ---------------------------------------------------------------------------
struct WaveBuf {
  alignas(32) unsigned zbufA[4 * 32 * 8];  // GEMM1 A operand, bf16 pairs, 4 K-tiles
  alignas(16) float    hbuf[16 * 256];     // raw layer-1 output (pre-LN)
  alignas(16) float    kout[16 * 128];     // raw layer-2 output (pre-LN)
  float statsH[32];                        // mu/rstd per row (layer 1)
  float statsK[32];                        // mu/rstd per row (layer 2)
  float tbuf[16];                          // per-row time argument for this stage
};

// Pack one 16x128 f32 tile (driver lane layout: lane L owns row L>>1,
// cols (L&1)*64 .. +63) into the WMMA 16-bit A layout as bf16 pairs.
__device__ __forceinline__ void pack_row_to_A(unsigned* zbufA, int lane,
                                              const float* za) {
  const int r = lane >> 1, cs = lane & 1;
#pragma unroll
  for (int ktl = 0; ktl < 2; ++ktl)
#pragma unroll
    for (int hi = 0; hi < 2; ++hi)
#pragma unroll
      for (int hf = 0; hf < 2; ++hf) {
        const int offb = 32 * ktl + 16 * hi + 8 * hf;
        uint4 v;
        v.x = pk_bf16(za[offb + 0], za[offb + 1]);
        v.y = pk_bf16(za[offb + 2], za[offb + 3]);
        v.z = pk_bf16(za[offb + 4], za[offb + 5]);
        v.w = pk_bf16(za[offb + 6], za[offb + 7]);
        const int ktA   = 2 * cs + ktl;   // global column / 32
        const int laneA = r + 16 * hf;    // A-layout lane holding this row/half
        *(uint4*)&zbufA[(ktA * 32 + laneA) * 8 + 4 * hi] = v;
      }
}

// One evaluation of f(z,t): GEMM1(+time rank-1 +bias) -> LN stats ->
// (LN+leaky fused into A2 pack) -> GEMM2(+bias) -> LN stats.
// N-tiles processed in pairs (two independent accumulator chains) so the
// bf16 WMMA->VALU hazard window of one tile is filled by the other tile's
// WMMAs / ds_loads instead of v_nops.
__device__ __forceinline__ void feval(
    WaveBuf& W, int lane, const unsigned* w1b, const unsigned* w2b,
    const float* w1t, const float* b1s, const float* g1s, const float* be1s,
    const float* b2s) {
  const int m16 = lane & 15;
  const int lh  = lane >> 4;

  v16bf A1[4];
#pragma unroll
  for (int kt = 0; kt < 4; ++kt)
    A1[kt] = *(const v16bf*)&W.zbufA[(kt * 32 + lane) * 8];

  float treg[8];
#pragma unroll
  for (int i = 0; i < 8; ++i) treg[i] = W.tbuf[i + 8 * lh];

  // ---- GEMM1: [16,128] x [128,256], two N-tiles in flight ----
#pragma unroll
  for (int nt = 0; nt < 16; nt += 2) {
    v8f accA = {}, accB = {};
#pragma unroll
    for (int kt = 0; kt < 4; ++kt) {
      const v16bf* bpA = (const v16bf*)&w1b[((kt * 16 + nt) * 32 + lane) * 8];
      const v16bf* bpB =
          (const v16bf*)&w1b[((kt * 16 + nt + 1) * 32 + lane) * 8];
      accA = wmma_bf16(A1[kt], *bpA, accA);
      accB = wmma_bf16(A1[kt], *bpB, accB);
    }
    const int nA = nt * 16 + m16;
    const int nB = nA + 16;
    const float wtA = w1t[nA], bbA = b1s[nA];
    const float wtB = w1t[nB], bbB = b1s[nB];
#pragma unroll
    for (int i = 0; i < 8; ++i) {
      W.hbuf[(i + 8 * lh) * 256 + nA] = accA[i] + treg[i] * wtA + bbA;
      W.hbuf[(i + 8 * lh) * 256 + nB] = accB[i] + treg[i] * wtB + bbB;
    }
  }

  // ---- LN stats over 256 features ----
  {
    const int r = lane >> 1, cb = (lane & 1) * 128;
    float s = 0.f, s2 = 0.f;
#pragma unroll
    for (int j = 0; j < 32; ++j) {
      float4 v = *(const float4*)&W.hbuf[r * 256 + cb + 4 * j];
      s  += v.x + v.y + v.z + v.w;
      s2 += v.x * v.x + v.y * v.y + v.z * v.z + v.w * v.w;
    }
    s  += __shfl_xor(s, 1, 32);
    s2 += __shfl_xor(s2, 1, 32);
    if ((lane & 1) == 0) {
      const float mu  = s * (1.f / 256.f);
      const float var = s2 * (1.f / 256.f) - mu * mu;
      W.statsH[2 * r]     = mu;
      W.statsH[2 * r + 1] = rsqrtf(var + LN_EPS);
    }
  }

  // ---- pack A2 with LN + leaky fused ----
  v16bf A2[8];
  {
    const float mu = W.statsH[2 * m16], rs = W.statsH[2 * m16 + 1];
#pragma unroll
    for (int kt = 0; kt < 8; ++kt) {
      V16U u;
#pragma unroll
      for (int d = 0; d < 8; ++d) {
        const int k = 32 * kt + 16 * (d >> 2) + 8 * lh + 2 * (d & 3);
        float2 h2 = *(const float2*)&W.hbuf[m16 * 256 + k];
        const float f0 = leaky((h2.x - mu) * rs * g1s[k]     + be1s[k]);
        const float f1 = leaky((h2.y - mu) * rs * g1s[k + 1] + be1s[k + 1]);
        u.u[d] = pk_bf16(f0, f1);
      }
      A2[kt] = u.v;
    }
  }

  // ---- GEMM2: [16,256] x [256,128], two N-tiles in flight ----
#pragma unroll
  for (int nt = 0; nt < 8; nt += 2) {
    v8f accA = {}, accB = {};
#pragma unroll
    for (int kt = 0; kt < 8; ++kt) {
      const v16bf* bpA = (const v16bf*)&w2b[((kt * 8 + nt) * 32 + lane) * 8];
      const v16bf* bpB =
          (const v16bf*)&w2b[((kt * 8 + nt + 1) * 32 + lane) * 8];
      accA = wmma_bf16(A2[kt], *bpA, accA);
      accB = wmma_bf16(A2[kt], *bpB, accB);
    }
    const int nA = nt * 16 + m16;
    const int nB = nA + 16;
    const float bbA = b2s[nA], bbB = b2s[nB];
#pragma unroll
    for (int i = 0; i < 8; ++i) {
      W.kout[(i + 8 * lh) * 128 + nA] = accA[i] + bbA;
      W.kout[(i + 8 * lh) * 128 + nB] = accB[i] + bbB;
    }
  }

  // ---- LN stats over 128 features ----
  {
    const int r = lane >> 1, cb = (lane & 1) * 64;
    float s = 0.f, s2 = 0.f;
#pragma unroll
    for (int j = 0; j < 16; ++j) {
      float4 v = *(const float4*)&W.kout[r * 128 + cb + 4 * j];
      s  += v.x + v.y + v.z + v.w;
      s2 += v.x * v.x + v.y * v.y + v.z * v.z + v.w * v.w;
    }
    s  += __shfl_xor(s, 1, 32);
    s2 += __shfl_xor(s2, 1, 32);
    if ((lane & 1) == 0) {
      const float mu  = s * (1.f / 128.f);
      const float var = s2 * (1.f / 128.f) - mu * mu;
      W.statsK[2 * r]     = mu;
      W.statsK[2 * r + 1] = rsqrtf(var + LN_EPS);
    }
  }
}

// RK4 stage s (0..3): apply LN+leaky to layer-2 output -> derivative k,
// accumulate kacc, build next z-argument (or final z update) + time arg.
__device__ __forceinline__ void driver_stage(
    int s, WaveBuf& W, int lane, float* z, float* kacc, float dt, float t_cur,
    float t_next, const float* g2s, const float* be2s, float* zrow) {
  const int r = lane >> 1, cs = lane & 1, c0 = cs * 64;
  const float mu = W.statsK[2 * r], rs = W.statsK[2 * r + 1];
  const float wgt = (s == 1 || s == 2) ? 2.f : 1.f;   // RK4 weights 1,2,2,1
  const float an  = (s == 2) ? 1.f : 0.5f;            // next-stage z coefficient
  float za[64];
#pragma unroll
  for (int off = 0; off < 64; off += 2) {
    const int c = c0 + off;
    float2 kr = *(const float2*)&W.kout[r * 128 + c];
    const float k0 = leaky((kr.x - mu) * rs * g2s[c]     + be2s[c]);
    const float k1 = leaky((kr.y - mu) * rs * g2s[c + 1] + be2s[c + 1]);
    if (s == 0) { kacc[off] = k0; kacc[off + 1] = k1; }
    else        { kacc[off] += wgt * k0; kacc[off + 1] += wgt * k1; }
    if (s < 3) {
      za[off]     = z[off]     + an * dt * k0;
      za[off + 1] = z[off + 1] + an * dt * k1;
    } else {
      z[off]     += (dt * (1.f / 6.f)) * kacc[off];
      z[off + 1] += (dt * (1.f / 6.f)) * kacc[off + 1];
      za[off] = z[off]; za[off + 1] = z[off + 1];
    }
  }
  pack_row_to_A(W.zbufA, lane, za);
  if (cs == 0) W.tbuf[r] = (s < 2) ? (t_cur + 0.5f * dt) : t_next;
  if (s == 3) {
#pragma unroll
    for (int q = 0; q < 16; ++q) {
      float4 v;
      v.x = z[4 * q]; v.y = z[4 * q + 1]; v.z = z[4 * q + 2]; v.w = z[4 * q + 3];
      *(float4*)&zrow[c0 + 4 * q] = v;
    }
  }
}

// ---------------------------------------------------------------------------
// ODE kernel: each wave integrates a 16-row batch tile for all T steps.
// Weights pre-swizzled into LDS in WMMA B layout (bf16 pairs).
// ---------------------------------------------------------------------------
__global__ __launch_bounds__(128, 1) void node_ode_kernel(
    const float* __restrict__ z0, const float* __restrict__ ts,
    const float* __restrict__ W1, const float* __restrict__ b1,
    const float* __restrict__ g1, const float* __restrict__ be1,
    const float* __restrict__ W2, const float* __restrict__ b2,
    const float* __restrict__ g2, const float* __restrict__ be2,
    float* __restrict__ zs) {
  __shared__ alignas(32) unsigned w1b[4 * 16 * 32 * 8];  // 64KB
  __shared__ alignas(32) unsigned w2b[8 * 8 * 32 * 8];   // 64KB
  __shared__ float w1t[256], b1s[256], g1s[256], be1s[256];
  __shared__ float b2s[128], g2s[128], be2s[128];
  __shared__ WaveBuf wb[4];

  const int tid = threadIdx.x;

  // Swizzle W1 (rows 0..127) into bf16-pair B layout.
  for (int s = tid; s < 4 * 16 * 32 * 8; s += 128) {
    const int d = s & 7, ln = (s >> 3) & 31, nt = (s >> 8) & 15, kt = s >> 12;
    const int k = 32 * kt + 16 * (d >> 2) + 8 * (ln >> 4) + 2 * (d & 3);
    const int n = 16 * nt + (ln & 15);
    w1b[s] = pk_bf16(W1[k * 256 + n], W1[(k + 1) * 256 + n]);
  }
  // Swizzle W2.
  for (int s = tid; s < 8 * 8 * 32 * 8; s += 128) {
    const int d = s & 7, ln = (s >> 3) & 31, nt = (s >> 8) & 7, kt = s >> 11;
    const int k = 32 * kt + 16 * (d >> 2) + 8 * (ln >> 4) + 2 * (d & 3);
    const int n = 16 * nt + (ln & 15);
    w2b[s] = pk_bf16(W2[k * 128 + n], W2[(k + 1) * 128 + n]);
  }
  for (int i = tid; i < 256; i += 128) {
    w1t[i] = W1[128 * 256 + i];  // time row of W1 (rank-1 epilogue)
    b1s[i] = b1[i]; g1s[i] = g1[i]; be1s[i] = be1[i];
  }
  for (int i = tid; i < 128; i += 128) {
    b2s[i] = b2[i]; g2s[i] = g2[i]; be2s[i] = be2[i];
  }
  __syncthreads();

  const int wave = tid >> 5, lane = tid & 31;
  WaveBuf& W = wb[wave];
  const int tileBase = (blockIdx.x * 4 + wave) * 16;
  const int r = lane >> 1, cs = lane & 1, c0 = cs * 64;
  const int brow = tileBase + r;

  float z[64], kacc[64];
#pragma unroll
  for (int q = 0; q < 16; ++q) {
    float4 v = *(const float4*)&z0[(size_t)brow * 128 + c0 + 4 * q];
    z[4 * q] = v.x; z[4 * q + 1] = v.y; z[4 * q + 2] = v.z; z[4 * q + 3] = v.w;
  }
  pack_row_to_A(W.zbufA, lane, z);
  if (cs == 0) W.tbuf[r] = 0.f;
  float t_cur = 0.f;

#pragma unroll 1
  for (int step = 0; step < TSTEPS; ++step) {
    const float t_next = ts[step * BATCH + brow];
    const float dt = t_next - t_cur;
    float* zrow = &zs[((size_t)step * BATCH + brow) * 128];
#pragma unroll 1
    for (int s = 0; s < 4; ++s) {
      feval(W, lane, w1b, w2b, w1t, b1s, g1s, be1s, b2s);
      driver_stage(s, W, lane, z, kacc, dt, t_cur, t_next, g2s, be2s, zrow);
    }
    t_cur = t_next;
  }
}

// ---------------------------------------------------------------------------
// Decoder kernel: xs = leaky(zs @ Wl + bl) @ Wo + bo   (Wo N padded to 16)
// ---------------------------------------------------------------------------
__global__ __launch_bounds__(128, 1) void node_decoder_kernel(
    const float* __restrict__ zs, const float* __restrict__ Wl,
    const float* __restrict__ bl, const float* __restrict__ Wo,
    const float* __restrict__ bo, float* __restrict__ xs) {
  __shared__ alignas(32) unsigned wlb[4 * 16 * 32 * 8];  // 64KB
  __shared__ alignas(32) unsigned wob[8 * 32 * 8];       // 8KB (N padded)
  __shared__ float bls[256], bos[16];
  __shared__ alignas(16) float hb[4][16 * 256];

  const int tid = threadIdx.x;
  for (int s = tid; s < 4 * 16 * 32 * 8; s += 128) {
    const int d = s & 7, ln = (s >> 3) & 31, nt = (s >> 8) & 15, kt = s >> 12;
    const int k = 32 * kt + 16 * (d >> 2) + 8 * (ln >> 4) + 2 * (d & 3);
    const int n = 16 * nt + (ln & 15);
    wlb[s] = pk_bf16(Wl[k * 256 + n], Wl[(k + 1) * 256 + n]);
  }
  for (int s = tid; s < 8 * 32 * 8; s += 128) {
    const int d = s & 7, ln = (s >> 3) & 31, kt = s >> 8;
    const int k = 32 * kt + 16 * (d >> 2) + 8 * (ln >> 4) + 2 * (d & 3);
    const int n = ln & 15;
    const float a = (n < OUTD) ? Wo[k * OUTD + n] : 0.f;
    const float b = (n < OUTD) ? Wo[(k + 1) * OUTD + n] : 0.f;
    wob[s] = pk_bf16(a, b);
  }
  for (int i = tid; i < 256; i += 128) bls[i] = bl[i];
  if (tid < 16) bos[tid] = (tid < OUTD) ? bo[tid] : 0.f;
  __syncthreads();

  const int wave = tid >> 5, lane = tid & 31;
  const int m16 = lane & 15, lh = lane >> 4;
  float* hbuf = hb[wave];

  const int nTiles = (TSTEPS * BATCH) / 16;
  const int stride = gridDim.x * 4;
#pragma unroll 1
  for (int tile = blockIdx.x * 4 + wave; tile < nTiles; tile += stride) {
    const size_t rowBase = (size_t)tile * 16;
    const size_t row = rowBase + m16;

    // A pack straight from global zs (f32 -> bf16 pairs)
    v16bf A1[4];
#pragma unroll
    for (int kt = 0; kt < 4; ++kt) {
      V16U u;
#pragma unroll
      for (int d = 0; d < 8; ++d) {
        const int k = 32 * kt + 16 * (d >> 2) + 8 * lh + 2 * (d & 3);
        float2 zv = *(const float2*)&zs[row * 128 + k];
        u.u[d] = pk_bf16(zv.x, zv.y);
      }
      A1[kt] = u.v;
    }

    // GEMM: [16,128] x [128,256], fused bias + leaky, two N-tiles in flight
#pragma unroll
    for (int nt = 0; nt < 16; nt += 2) {
      v8f accA = {}, accB = {};
#pragma unroll
      for (int kt = 0; kt < 4; ++kt) {
        const v16bf* bpA =
            (const v16bf*)&wlb[((kt * 16 + nt) * 32 + lane) * 8];
        const v16bf* bpB =
            (const v16bf*)&wlb[((kt * 16 + nt + 1) * 32 + lane) * 8];
        accA = wmma_bf16(A1[kt], *bpA, accA);
        accB = wmma_bf16(A1[kt], *bpB, accB);
      }
      const int nA = nt * 16 + m16;
      const int nB = nA + 16;
      const float bbA = bls[nA], bbB = bls[nB];
#pragma unroll
      for (int i = 0; i < 8; ++i) {
        hbuf[(i + 8 * lh) * 256 + nA] = leaky(accA[i] + bbA);
        hbuf[(i + 8 * lh) * 256 + nB] = leaky(accB[i] + bbB);
      }
    }

    // A2 pack from LDS
    v16bf A2[8];
#pragma unroll
    for (int kt = 0; kt < 8; ++kt) {
      V16U u;
#pragma unroll
      for (int d = 0; d < 8; ++d) {
        const int k = 32 * kt + 16 * (d >> 2) + 8 * lh + 2 * (d & 3);
        float2 h2 = *(const float2*)&hbuf[m16 * 256 + k];
        u.u[d] = pk_bf16(h2.x, h2.y);
      }
      A2[kt] = u.v;
    }

    // GEMM: [16,256] x [256,16(pad)]
    v8f acc = {};
#pragma unroll
    for (int kt = 0; kt < 8; ++kt) {
      const v16bf* bp = (const v16bf*)&wob[(kt * 32 + lane) * 8];
      acc = wmma_bf16(A2[kt], *bp, acc);
    }
    if (m16 < OUTD) {
      const float bb = bos[m16];
#pragma unroll
      for (int i = 0; i < 8; ++i)
        xs[(rowBase + i + 8 * lh) * OUTD + m16] = acc[i] + bb;
    }
  }
}

extern "C" void kernel_launch(void* const* d_in, const int* in_sizes, int n_in,
                              void* d_out, int out_size, void* d_ws,
                              size_t ws_size, hipStream_t stream) {
  const float* z0  = (const float*)d_in[0];
  const float* ts  = (const float*)d_in[1];
  const float* W1  = (const float*)d_in[2];
  const float* b1  = (const float*)d_in[3];
  const float* g1  = (const float*)d_in[4];
  const float* be1 = (const float*)d_in[5];
  const float* W2  = (const float*)d_in[6];
  const float* b2  = (const float*)d_in[7];
  const float* g2  = (const float*)d_in[8];
  const float* be2 = (const float*)d_in[9];
  const float* Wl  = (const float*)d_in[10];
  const float* bl  = (const float*)d_in[11];
  const float* Wo  = (const float*)d_in[12];
  const float* bo  = (const float*)d_in[13];

  float* xs = (float*)d_out;                          // [T,B,8] first
  float* zs = xs + (size_t)TSTEPS * BATCH * OUTD;     // then [T,B,128]

  // 1024 batch tiles of 16 rows, 4 waves/WG -> 256 workgroups
  node_ode_kernel<<<dim3(BATCH / 64), dim3(128), 0, stream>>>(
      z0, ts, W1, b1, g1, be1, W2, b2, g2, be2, zs);
  // 32768 row tiles, grid-stride over 512 WGs x 4 waves
  node_decoder_kernel<<<dim3(512), dim3(128), 0, stream>>>(zs, Wl, bl, Wo, bo,
                                                           xs);
}
